// Block_15066745274698
// MI455X (gfx1250) — compile-verified, matrix operand order checked by reading, not run on hardware
//
#include <hip/hip_runtime.h>
#include <hip/hip_bf16.h>

// ---------------------------------------------------------------------------
// Problem constants (from reference)
// ---------------------------------------------------------------------------
constexpr int T       = 4096;
constexpr int C       = 2048;
constexpr int NH      = 16;
constexpr int HS      = 128;
constexpr int NQG     = 4;
constexpr int QPKV    = NH / NQG;            // 4
constexpr int QKV_DIM = (NH + 2 * NQG) * HS; // 3072
constexpr int ROT     = HS / 2;              // 64

typedef __attribute__((ext_vector_type(8)))  _Float16 v8h;
typedef __attribute__((ext_vector_type(16))) _Float16 v16h;
typedef __attribute__((ext_vector_type(8)))  float    v8f;

// ---------------------------------------------------------------------------
// WMMA helpers
// ---------------------------------------------------------------------------
// CDNA5 16-bit A/B fragment layout (CDNA5 ISA):
//   lane L (<16):  M(or N)=L,  K = {0..7, 16..23}
//   lane L+16:     M(or N)=L,  K = {8..15, 24..31}
// => per lane: two contiguous 8-half (16B) chunks, the second 16 halfs ahead.
// `p` must already point at  row_base + kchunk + 8*(lane>>4).
static __device__ inline v16h load_frag(const _Float16* p) {
    v8h lo = *(const v8h*)(p);
    v8h hi = *(const v8h*)(p + 16);
    return __builtin_shufflevector(lo, hi, 0, 1, 2, 3, 4, 5, 6, 7,
                                           8, 9, 10, 11, 12, 13, 14, 15);
}

static __device__ inline v8f wmma_f16(v16h a, v16h b, v8f c) {
    // (neg_a, A, neg_b, B, c_mod, C, reuse_a, reuse_b)
    return __builtin_amdgcn_wmma_f32_16x16x32_f16(false, a, false, b,
                                                  (short)0, c, false, false);
}

// ---------------------------------------------------------------------------
// CDNA5 async global->LDS copy (ASYNCcnt path, CDNA5 ISA).
// Inline asm is portable across ROCm-7.2 / amdgpu-toolchain builtin skews.
//   GLOBAL_LOAD_ASYNC_TO_LDS_B128 vdst(=LDS byte addr), vaddr(64b), off
// Generic pointers to LDS carry the LDS byte offset in addr[31:0] (ISA 10.2).
// ---------------------------------------------------------------------------
static __device__ inline void async_g2l_b128(unsigned lds_addr, const void* gaddr) {
    asm volatile("global_load_async_to_lds_b128 %0, %1, off"
                 :: "v"(lds_addr), "v"(gaddr)
                 : "memory");
}

static __device__ inline void wait_async0() {
#if __has_builtin(__builtin_amdgcn_s_wait_asynccnt)
    __builtin_amdgcn_s_wait_asynccnt(0);
#else
    asm volatile("s_wait_asynccnt 0x0" ::: "memory");
#endif
}

// ---------------------------------------------------------------------------
// f32 -> f16 conversion (grid-stride)
// ---------------------------------------------------------------------------
__global__ void cvt_f32_to_f16(const float* __restrict__ src,
                               _Float16* __restrict__ dst, int n) {
    for (int i = blockIdx.x * blockDim.x + threadIdx.x; i < n;
         i += gridDim.x * blockDim.x)
        dst[i] = (_Float16)src[i];
}

// ---------------------------------------------------------------------------
// GEMM:  Cout[M,N] = A[M,K] * Bt[N,K]^T   (f16 in, f32 out)
// 8 waves / block; wave w owns rows [blockRow + 16w, +16), 64 columns.
// The 64x32 B tile is shared by all 8 waves -> staged once per block in LDS
// via async global->LDS b128 copies, double-buffered so the next tile
// streams in behind the WMMAs. B rows padded to 40 halfs (80 B) so the
// per-fragment ds_load_b128 pattern is bank-conflict-free. All four B
// fragments are fetched into registers before any WMMA issues, so the LDS
// loads clause up and the matrix pipe doesn't eat per-fragment dscnt waits.
// ---------------------------------------------------------------------------
constexpr int BSTRIDE = 40;  // halfs per LDS B row (32 data + 8 pad)

__global__ void __launch_bounds__(256)
gemm_f16f32(const _Float16* __restrict__ A, const _Float16* __restrict__ Bt,
            float* __restrict__ Cout, int M, int N, int K) {
    const int tid  = threadIdx.x;
    const int wave = tid >> 5;
    const int lane = tid & 31;
    const int lh   = lane >> 4;   // half-wave selector
    const int ln   = lane & 15;

    const int row0 = blockIdx.y * 128 + wave * 16;
    const int col0 = blockIdx.x * 64;
    const int m    = row0 + ln;

    __shared__ __align__(16) _Float16 bbuf[2][64 * BSTRIDE];

    // Cooperative B-tile fetch mapping: 256 threads x 16 B == 64x32 halfs.
    const int brow   = tid >> 2;  // 0..63  (B tile row = output column)
    const int bchunk = tid & 3;   // 0..3   (8-half chunk within the row)
    const _Float16* gsrc = Bt + (size_t)(col0 + brow) * K + bchunk * 8;
    const unsigned ldsoff0 =
        (unsigned)(uintptr_t)(&bbuf[0][brow * BSTRIDE + bchunk * 8]);
    const unsigned ldsoff1 =
        (unsigned)(uintptr_t)(&bbuf[1][brow * BSTRIDE + bchunk * 8]);

    v8f acc[4];
#pragma unroll
    for (int j = 0; j < 4; ++j)
#pragma unroll
        for (int r = 0; r < 8; ++r) acc[j][r] = 0.0f;

    const _Float16* arow = A + (size_t)m * K + lh * 8;
    const int nk = K / 32;

    // Prologue: stream first B tile into buffer 0.
    async_g2l_b128(ldsoff0, gsrc);

    for (int ki = 0; ki < nk; ++ki) {
        const int cur = ki & 1;

        wait_async0();       // our slice of buf[cur] landed
        __syncthreads();     // everyone's slice landed; prev reads retired

        if (ki + 1 < nk)     // stream next tile behind the WMMAs
            async_g2l_b128(cur ? ldsoff0 : ldsoff1,
                           gsrc + (size_t)(ki + 1) * 32);

        v16h a = load_frag(arow + ki * 32);
        const _Float16* bbase = &bbuf[cur][0];

        // Fetch all four B fragments first (one ds clause), then fire the
        // four WMMAs back-to-back.
        v16h b[4];
#pragma unroll
        for (int j = 0; j < 4; ++j)
            b[j] = load_frag(bbase + (j * 16 + ln) * BSTRIDE + lh * 8);
#pragma unroll
        for (int j = 0; j < 4; ++j)
            acc[j] = wmma_f16(a, b[j], acc[j]);
    }

#pragma unroll
    for (int j = 0; j < 4; ++j)
#pragma unroll
        for (int r = 0; r < 8; ++r)
            Cout[(size_t)(row0 + r + 8 * lh) * N + col0 + j * 16 + ln] = acc[j][r];
}

// ---------------------------------------------------------------------------
// RoPE + split qkv[T, 3072] (f32) into:
//   Qf[NH][T][HS]  (f16, rotated)
//   Kf[NQG][T][HS] (f16, rotated)
//   Vt[NQG][HS][T] (f16, TRANSPOSED so P*V B-frags are key-contiguous)
// ---------------------------------------------------------------------------
__global__ void rope_split(const float* __restrict__ qkv,
                           const float* __restrict__ cs,
                           const float* __restrict__ sn,
                           _Float16* __restrict__ Qf,
                           _Float16* __restrict__ Kf,
                           _Float16* __restrict__ Vt) {
    int i = blockIdx.x * blockDim.x + threadIdx.x;
    if (i >= T * QKV_DIM) return;
    int t = i / QKV_DIM;
    int f = i - t * QKV_DIM;
    int g   = f / ((QPKV + 2) * HS);
    int rem = f - g * (QPKV + 2) * HS;
    int j   = rem / HS;
    int d   = rem - j * HS;

    float v = qkv[(size_t)t * QKV_DIM + f];
    if (j == QPKV + 1) {                       // V: plain convert, transposed
        Vt[((size_t)(g * HS + d)) * T + t] = (_Float16)v;
        return;
    }
    // RoPE (q heads and k head)
    int   d2 = (d < ROT) ? d : d - ROT;
    float c  = cs[t * ROT + d2];
    float s  = sn[t * ROT + d2];
    int   base = f - d;                        // start of this head's 128 dims
    float partner = qkv[(size_t)t * QKV_DIM + base + ((d < ROT) ? d + ROT : d - ROT)];
    float o = (d < ROT) ? (v * c - partner * s) : (v * c + partner * s);

    if (j < QPKV) {
        int h = g * QPKV + j;
        Qf[((size_t)h * T + t) * HS + d] = (_Float16)o;
    } else {
        Kf[((size_t)g * T + t) * HS + d] = (_Float16)o;
    }
}

// ---------------------------------------------------------------------------
// Flash attention (causal, GQA). Grid: (T/64, NH). Block: 128 thr = 4 waves.
// Each wave owns a 16-query tile; streams keys in blocks of 32; online
// softmax; LDS staging to convert the f32 P tile into the f16 A-fragment
// layout for the P*V WMMA. Y written as [T, C] f16 so it feeds the
// projection GEMM directly.
// ---------------------------------------------------------------------------
__global__ void __launch_bounds__(128)
attn_kernel(const _Float16* __restrict__ Qf, const _Float16* __restrict__ Kf,
            const _Float16* __restrict__ Vt, _Float16* __restrict__ Y) {
    const int wave = threadIdx.x >> 5;
    const int lane = threadIdx.x & 31;
    const int lh   = lane >> 4;
    const int ln   = lane & 15;

    const int h   = blockIdx.y;
    const int hkv = h / QPKV;
    const int qb  = blockIdx.x * 64 + wave * 16;

    const _Float16* Qh = Qf + ((size_t)h * T) * HS;
    const _Float16* Kp = Kf + ((size_t)hkv * T) * HS;
    const _Float16* Vp = Vt + ((size_t)hkv * HS) * T;

    // Persistent Q fragments: 16 rows x 128 dims = 4 k-chunks
    v16h qf[4];
    {
        const _Float16* qrow = Qh + (size_t)(qb + ln) * HS + lh * 8;
#pragma unroll
        for (int kc = 0; kc < 4; ++kc) qf[kc] = load_frag(qrow + kc * 32);
    }

    v8f acc[8];
#pragma unroll
    for (int cI = 0; cI < 8; ++cI)
#pragma unroll
        for (int r = 0; r < 8; ++r) acc[cI][r] = 0.0f;

    float mrow[8], lrow[8];
#pragma unroll
    for (int r = 0; r < 8; ++r) { mrow[r] = -1.0e30f; lrow[r] = 0.0f; }

    __shared__ __align__(16) _Float16 pbuf[4][16 * 32];   // per-wave P tile
    _Float16* pb = &pbuf[wave][0];

    const float sc = 0.08838834764831845f;  // 1/sqrt(128)

    // Uniform trip count across the block (legal __syncthreads); tail blocks
    // for low-qb waves are fully masked and contribute exactly zero.
    const int nkb = (blockIdx.x * 64 + 64) >> 5;
    for (int kb = 0; kb < nkb; ++kb) {
        const int s0 = kb * 32;

        // ---- S = Q * K^T  (two 16x16 f32 tiles for 32 keys) ----
        v8f S0, S1;
#pragma unroll
        for (int r = 0; r < 8; ++r) { S0[r] = 0.0f; S1[r] = 0.0f; }
        {
            const _Float16* k0 = Kp + (size_t)(s0 + ln) * HS + lh * 8;
            const _Float16* k1 = Kp + (size_t)(s0 + 16 + ln) * HS + lh * 8;
#pragma unroll
            for (int kc = 0; kc < 4; ++kc) {
                v16h b0 = load_frag(k0 + kc * 32);
                v16h b1 = load_frag(k1 + kc * 32);
                S0 = wmma_f16(qf[kc], b0, S0);
                S1 = wmma_f16(qf[kc], b1, S1);
            }
        }

        // ---- scale, causal mask, online softmax stats ----
#pragma unroll
        for (int r = 0; r < 8; ++r) {
            const int trow = qb + r + 8 * lh;
            float a0 = S0[r] * sc;
            float a1 = S1[r] * sc;
            if (s0 + ln > trow)      a0 = -1.0e30f;
            if (s0 + 16 + ln > trow) a1 = -1.0e30f;

            float mx = fmaxf(a0, a1);
#pragma unroll
            for (int off = 1; off < 16; off <<= 1)
                mx = fmaxf(mx, __shfl_xor(mx, off, 32));
            const float mnew = fmaxf(mrow[r], mx);

            const float p0 = __expf(a0 - mnew);
            const float p1 = __expf(a1 - mnew);
            S0[r] = p0;
            S1[r] = p1;

            float rs = p0 + p1;
#pragma unroll
            for (int off = 1; off < 16; off <<= 1)
                rs += __shfl_xor(rs, off, 32);

            const float esc = __expf(mrow[r] - mnew);
            lrow[r] = lrow[r] * esc + rs;
            mrow[r] = mnew;
#pragma unroll
            for (int cI = 0; cI < 8; ++cI) acc[cI][r] *= esc;
        }

        // ---- stage P (C-layout f32 -> row-major f16 in LDS) ----
        __syncthreads();
#pragma unroll
        for (int r = 0; r < 8; ++r) {
            const int M = r + 8 * lh;
            pb[M * 32 + ln]      = (_Float16)S0[r];
            pb[M * 32 + 16 + ln] = (_Float16)S1[r];
        }
        __syncthreads();

        // ---- reload P in A-fragment layout, then O += P * V ----
        v16h pf = load_frag(pb + ln * 32 + lh * 8);
#pragma unroll
        for (int cI = 0; cI < 8; ++cI) {
            const _Float16* vrow = Vp + (size_t)(cI * 16 + ln) * T + s0 + lh * 8;
            v16h bv = load_frag(vrow);
            acc[cI] = wmma_f16(pf, bv, acc[cI]);
        }
    }

    // ---- normalize and store Y[T, C] ----
#pragma unroll
    for (int r = 0; r < 8; ++r) {
        const float inv = 1.0f / lrow[r];
        const size_t trow = (size_t)(qb + r + 8 * lh);
#pragma unroll
        for (int cI = 0; cI < 8; ++cI)
            Y[trow * C + h * HS + cI * 16 + ln] = (_Float16)(acc[cI][r] * inv);
    }
}

// ---------------------------------------------------------------------------
// Host-side launch
// ---------------------------------------------------------------------------
extern "C" void kernel_launch(void* const* d_in, const int* in_sizes, int n_in,
                              void* d_out, int out_size, void* d_ws, size_t ws_size,
                              hipStream_t stream) {
    (void)in_sizes; (void)n_in; (void)out_size; (void)ws_size;

    const float* x      = (const float*)d_in[0];   // [1, T, C]
    const float* cosT   = (const float*)d_in[1];   // [T, ROT]
    const float* sinT   = (const float*)d_in[2];   // [T, ROT]
    const float* W_attn = (const float*)d_in[3];   // [QKV_DIM, C]
    const float* W_proj = (const float*)d_in[4];   // [C, C]
    float* out = (float*)d_out;                    // [1, T, C]

    // Workspace carve-up (~124 MB)
    char* ws = (char*)d_ws;
    _Float16* xh  = (_Float16*)(ws);                             // 16 MB
    _Float16* Wah = (_Float16*)(ws + (size_t)16  * 1024 * 1024); // 12 MB
    _Float16* Wph = (_Float16*)(ws + (size_t)28  * 1024 * 1024); //  8 MB
    float*    qkv = (float*)   (ws + (size_t)36  * 1024 * 1024); // 48 MB
    _Float16* Qf  = (_Float16*)(ws + (size_t)84  * 1024 * 1024); // 16 MB
    _Float16* Kf  = (_Float16*)(ws + (size_t)100 * 1024 * 1024); //  4 MB
    _Float16* Vt  = (_Float16*)(ws + (size_t)104 * 1024 * 1024); //  4 MB
    _Float16* Yh  = (_Float16*)(ws + (size_t)108 * 1024 * 1024); // 16 MB

    // 1) f32 -> f16 conversions
    cvt_f32_to_f16<<<4096, 256, 0, stream>>>(x, xh, T * C);
    cvt_f32_to_f16<<<4096, 256, 0, stream>>>(W_attn, Wah, QKV_DIM * C);
    cvt_f32_to_f16<<<4096, 256, 0, stream>>>(W_proj, Wph, C * C);

    // 2) QKV projection: qkv[T, 3072] = xh * Wah^T
    {
        dim3 grid(QKV_DIM / 64, T / 128);
        gemm_f16f32<<<grid, 256, 0, stream>>>(xh, Wah, qkv, T, QKV_DIM, C);
    }

    // 3) RoPE + head split (+ V transpose)
    {
        int total = T * QKV_DIM;
        rope_split<<<(total + 255) / 256, 256, 0, stream>>>(qkv, cosT, sinT,
                                                            Qf, Kf, Vt);
    }

    // 4) Flash attention -> Yh[T, C] f16
    {
        dim3 grid(T / 64, NH);
        attn_kernel<<<grid, 128, 0, stream>>>(Qf, Kf, Vt, Yh);
    }

    // 5) Output projection: out[T, C] = Yh * Wph^T
    {
        dim3 grid(C / 64, T / 128);
        gemm_f16f32<<<grid, 256, 0, stream>>>(Yh, Wph, out, T, C, C);
    }
}
